// RCPNA_42812234006661
// MI455X (gfx1250) — matrix operand order repaired and checked by reading, not compiled, not run on hardware
//
#include <hip/hip_runtime.h>

// ---------------- problem constants (fixed by the reference) ----------------
#define BBATCH 16384
#define NNODE  11
#define DRAW   4
#define FDIM   32
#define EDIMC  4
#define NINT   3
#define EPSAMP 17                    // edges per sample (fixed DAG)
#define NTOT   (BBATCH*NNODE)        // 180224 nodes
#define ETOT   (BBATCH*EPSAMP)       // 278528 edges
#define KPRE   (2*FDIM+EDIMC)        // 68  (pre-MLP input dim)
#define KPREP  96                    // 68 padded to 3 chunks of K=32
#define KPOST  (12*FDIM)             // 384 (post-MLP input dim), 12 chunks
// avg_log = (4*ln2 + 6*ln3 + ln5)/11  (in-degrees of the fixed DAG)
#define AVG_LOG 0.9976091242438673f

// fixed topology of one sample's DAG
__constant__ int   c_src[EPSAMP]   = {0,0,1,1,2,0,3,4,5,6,7,2,8,3,6,9,1};
__constant__ int   c_dst[EPSAMP]   = {3,4,3,5,5,6,6,7,7,8,8,9,9,10,10,10,10};
__constant__ int   c_deg[NNODE]    = {0,0,0,2,1,2,2,2,2,2,4};
__constant__ int   c_eoff[NNODE]   = {0,0,0,0,2,3,5,7,9,11,13};
__constant__ int   c_elist[EPSAMP] = {0,2,1,3,4,5,6,7,8,9,10,11,12,13,14,15,16};
__constant__ float c_logd[NNODE]   = {0.6931471805599453f,0.6931471805599453f,0.6931471805599453f,
                                      1.0986122886681098f,0.6931471805599453f,1.0986122886681098f,
                                      1.0986122886681098f,1.0986122886681098f,1.0986122886681098f,
                                      1.0986122886681098f,1.6094379124341003f};

typedef _Float16 half8  __attribute__((ext_vector_type(8)));
typedef _Float16 half16 __attribute__((ext_vector_type(16)));
typedef float    float8 __attribute__((ext_vector_type(8)));

// load a 16-half WMMA fragment: halves 0..7 at p (K+0..7), 8..15 at p+16 (K+16..23)
__device__ __forceinline__ half16 ld_frag16(const _Float16* p) {
  half8 lo = *(const half8*)p;
  half8 hi = *(const half8*)(p + 16);
  return __builtin_shufflevector(lo, hi, 0,1,2,3,4,5,6,7,8,9,10,11,12,13,14,15);
}
// load a 16-half B fragment stored contiguously per lane
__device__ __forceinline__ half16 ld_bfrag16(const _Float16* p) {
  half8 lo = *(const half8*)p;
  half8 hi = *(const half8*)(p + 8);
  return __builtin_shufflevector(lo, hi, 0,1,2,3,4,5,6,7,8,9,10,11,12,13,14,15);
}
__device__ __forceinline__ float8 wmma16x16x32(half16 a, half16 b, float8 c) {
  return __builtin_amdgcn_wmma_f32_16x16x32_f16(false, a, false, b, (short)0, c, false, false);
}
// wave-local LDS produce->consume fence (no workgroup barrier needed)
__device__ __forceinline__ void wave_lds_fence() {
  __builtin_amdgcn_wave_barrier();
  asm volatile("s_wait_dscnt 0" ::: "memory");
  __builtin_amdgcn_wave_barrier();
}

// ---------------- encoder: H = relu(X @ W_emb[n] + b_emb[n]), stored f16 ----
__global__ void __launch_bounds__(256)
k_encoder(const float* __restrict__ x, const float* __restrict__ Wemb,
          const float* __restrict__ bemb, _Float16* __restrict__ H) {
  int gid = blockIdx.x * 256 + threadIdx.x;
  if (gid >= NTOT * FDIM) return;
  int f = gid & (FDIM - 1);
  int node = gid >> 5;
  int b = node / NNODE;
  int n = node - b * NNODE;
  const float* xp = x + b * (NNODE * DRAW) + n * DRAW;
  const float* wp = Wemb + n * DRAW * FDIM + f;
  float acc = bemb[n * FDIM + f];
  acc += xp[0] * wp[0] + xp[1] * wp[FDIM] + xp[2] * wp[2 * FDIM] + xp[3] * wp[3 * FDIM];
  H[gid] = (_Float16)(acc > 0.f ? acc : 0.f);
}

// ---------------- edge pre-MLP: m = relu([h_dst,h_src,ea] @ Wpre + b) -------
// block = 128 (4 waves), each wave owns one 16-edge tile at a time.
__global__ void __launch_bounds__(128)
k_edge_pre(const _Float16* __restrict__ Hin, const float* __restrict__ eattr,
           const float* __restrict__ Wpre, const float* __restrict__ bpre,
           float* __restrict__ mout) {
  __shared__ _Float16 sB[3 * 2 * 32 * 16];      // 6 KB, weight fragments
  __shared__ _Float16 sA[4][16 * KPREP];        // 4 x 3 KB, A rows per wave
  const int tid = threadIdx.x;

  // stage Wpre (68x32 f32 -> padded 96x32 f16) into per-lane fragment layout
  for (int idx = tid; idx < 3 * 2 * 32 * 16; idx += 128) {
    int j = idx & 15;
    int l = (idx >> 4) & 31;
    int t = (idx >> 9) & 1;
    int c = idx >> 10;
    int k = c * 32 + ((l >> 4) << 3) + (j & 7) + ((j >> 3) << 4);       // K index
    int col = t * 16 + (l & 15);                // N index
    float w = (k < KPRE) ? Wpre[k * FDIM + col] : 0.0f;
    sB[idx] = (_Float16)w;
  }
  __syncthreads();

  const int wv = tid >> 5;
  const int lane = tid & 31;
  const int r = lane & 15;
  const int part = lane >> 4;
  const int gw = blockIdx.x * 4 + wv;
  const int nw = gridDim.x * 4;
  _Float16* A = sA[wv];
  const int col0 = lane & 15;
  const float b0 = bpre[col0];
  const float b1 = bpre[16 + col0];

  for (int tile = gw; tile < ETOT / 16; tile += nw) {
    const int ge = tile * 16 + r;
    const int s = ge / EPSAMP;
    const int es = ge - s * EPSAMP;
    // build A row r: [h_dst(32) | h_src(32) | ea(4) | zeros(28)]  (2 lanes/row)
    if (part == 0) {
      int node = s * NNODE + c_dst[es];
      const int4* sp = (const int4*)(Hin + node * FDIM);
      int4* dp = (int4*)(A + r * KPREP);
      dp[0] = sp[0]; dp[1] = sp[1]; dp[2] = sp[2]; dp[3] = sp[3];
    } else {
      int node = s * NNODE + c_src[es];
      const int4* sp = (const int4*)(Hin + node * FDIM);
      int4* dp = (int4*)(A + r * KPREP + 32);
      dp[0] = sp[0]; dp[1] = sp[1]; dp[2] = sp[2]; dp[3] = sp[3];
      float4 ea = *(const float4*)(eattr + (size_t)ge * EDIMC);
      union { _Float16 h[8]; int4 v; } u;
      u.h[0] = (_Float16)ea.x; u.h[1] = (_Float16)ea.y;
      u.h[2] = (_Float16)ea.z; u.h[3] = (_Float16)ea.w;
      u.h[4] = (_Float16)0.f;  u.h[5] = (_Float16)0.f;
      u.h[6] = (_Float16)0.f;  u.h[7] = (_Float16)0.f;
      int4 z; z.x = z.y = z.z = z.w = 0;
      *(int4*)(A + r * KPREP + 64) = u.v;
      *(int4*)(A + r * KPREP + 72) = z;
      *(int4*)(A + r * KPREP + 80) = z;
      *(int4*)(A + r * KPREP + 88) = z;
    }
    wave_lds_fence();

    float8 acc0 = {}; float8 acc1 = {};
    #pragma unroll
    for (int c = 0; c < 3; ++c) {
      half16 af = ld_frag16(A + r * KPREP + c * 32 + part * 8);
      half16 bf0 = ld_bfrag16(sB + ((c * 2 + 0) * 32 + lane) * 16);
      half16 bf1 = ld_bfrag16(sB + ((c * 2 + 1) * 32 + lane) * 16);
      acc0 = wmma16x16x32(af, bf0, acc0);
      acc1 = wmma16x16x32(af, bf1, acc1);
    }
    const int rowbase = tile * 16 + part * 8;
    #pragma unroll
    for (int v = 0; v < 8; ++v) {
      float m0 = acc0[v] + b0; m0 = m0 > 0.f ? m0 : 0.f;
      float m1 = acc1[v] + b1; m1 = m1 > 0.f ? m1 : 0.f;
      mout[(size_t)(rowbase + v) * FDIM + col0] = m0;
      mout[(size_t)(rowbase + v) * FDIM + 16 + col0] = m1;
    }
    wave_lds_fence();   // reads done before next tile overwrites A
  }
}

// per-(node,feature) PNA stats from messages m
__device__ __forceinline__ void pna_stats(const float* __restrict__ m, int s, int n,
                                          int lane, float& mean, float& mn, float& mx,
                                          float& sd) {
  int d = c_deg[n];
  int eo = c_eoff[n];
  float sum = 0.f, ssq = 0.f;
  float lo = 3.4e38f, hi = -3.4e38f;
  for (int j = 0; j < d; ++j) {
    float v = m[(size_t)(s * EPSAMP + c_elist[eo + j]) * FDIM + lane];
    sum += v; ssq += v * v;
    lo = fminf(lo, v); hi = fmaxf(hi, v);
  }
  float cc = d > 0 ? (float)d : 1.0f;
  mean = sum / cc;
  float var = ssq / cc - mean * mean;
  var = var > 0.f ? var : 0.f;
  sd = sqrtf(var + 1e-5f);
  mn = d > 0 ? lo : 0.f;
  mx = d > 0 ? hi : 0.f;
}

// ---------------- node post-MLP layer1: h1 = relu(scaled @ Wpost + b) -------
// block = 64 (2 waves); each wave owns one 16-node tile at a time.
__global__ void __launch_bounds__(64)
k_node_post1(const float* __restrict__ m, const float* __restrict__ Wpost,
             const float* __restrict__ bpost, _Float16* __restrict__ h1) {
  __shared__ _Float16 sB[12 * 2 * 32 * 16];     // 24 KB
  __shared__ _Float16 sA[2][16 * KPOST];        // 2 x 12 KB
  const int tid = threadIdx.x;
  for (int idx = tid; idx < 12 * 2 * 32 * 16; idx += 64) {
    int j = idx & 15;
    int l = (idx >> 4) & 31;
    int t = (idx >> 9) & 1;
    int c = idx >> 10;
    int k = c * 32 + ((l >> 4) << 3) + (j & 7) + ((j >> 3) << 4);
    int col = t * 16 + (l & 15);
    sB[idx] = (_Float16)Wpost[k * FDIM + col];
  }
  __syncthreads();

  const int wv = tid >> 5;
  const int lane = tid & 31;
  const int r = lane & 15;
  const int part = lane >> 4;
  const int gw = blockIdx.x * 2 + wv;
  const int nw = gridDim.x * 2;
  _Float16* A = sA[wv];
  const int col0 = lane & 15;
  const float b0 = bpost[col0];
  const float b1 = bpost[16 + col0];

  for (int tile = gw; tile < NTOT / 16; tile += nw) {
    // aggregation: iteration t = row, lane = feature (coalesced m reads)
    for (int t = 0; t < 16; ++t) {
      int gid = tile * 16 + t;
      int s = gid / NNODE;
      int n = gid - s * NNODE;
      float mean, mn, mx, sd;
      pna_stats(m, s, n, lane, mean, mn, mx, sd);
      float amp = c_logd[n] * (1.0f / AVG_LOG);
      float att = AVG_LOG / c_logd[n];
      _Float16* row = A + t * KPOST;
      row[0 * 32 + lane] = (_Float16)mean;
      row[1 * 32 + lane] = (_Float16)mn;
      row[2 * 32 + lane] = (_Float16)mx;
      row[3 * 32 + lane] = (_Float16)sd;
      row[4 * 32 + lane] = (_Float16)(mean * amp);
      row[5 * 32 + lane] = (_Float16)(mn * amp);
      row[6 * 32 + lane] = (_Float16)(mx * amp);
      row[7 * 32 + lane] = (_Float16)(sd * amp);
      row[8 * 32 + lane] = (_Float16)(mean * att);
      row[9 * 32 + lane] = (_Float16)(mn * att);
      row[10 * 32 + lane] = (_Float16)(mx * att);
      row[11 * 32 + lane] = (_Float16)(sd * att);
    }
    wave_lds_fence();

    float8 acc0 = {}; float8 acc1 = {};
    #pragma unroll
    for (int c = 0; c < 12; ++c) {
      half16 af = ld_frag16(A + r * KPOST + c * 32 + part * 8);
      half16 bf0 = ld_bfrag16(sB + ((c * 2 + 0) * 32 + lane) * 16);
      half16 bf1 = ld_bfrag16(sB + ((c * 2 + 1) * 32 + lane) * 16);
      acc0 = wmma16x16x32(af, bf0, acc0);
      acc1 = wmma16x16x32(af, bf1, acc1);
    }
    const int rowbase = tile * 16 + part * 8;
    #pragma unroll
    for (int v = 0; v < 8; ++v) {
      float v0 = acc0[v] + b0; v0 = v0 > 0.f ? v0 : 0.f;
      float v1 = acc1[v] + b1; v1 = v1 > 0.f ? v1 : 0.f;
      h1[(size_t)(rowbase + v) * FDIM + col0] = (_Float16)v0;
      h1[(size_t)(rowbase + v) * FDIM + 16 + col0] = (_Float16)v1;
    }
    wave_lds_fence();
  }
}

// ---------------- node post-MLP layer2 (out dim 1) + sigmoid + batch mean ---
__global__ void __launch_bounds__(256)
k_node_post2(const float* __restrict__ m, const float* __restrict__ W2,
             const float* __restrict__ b2p, float* __restrict__ pmout) {
  __shared__ float sW[KPOST];
  __shared__ float pmacc[NNODE];
  const int tid = threadIdx.x;
  if (tid < NNODE) pmacc[tid] = 0.f;
  for (int idx = tid; idx < KPOST; idx += 256) sW[idx] = W2[idx];
  __syncthreads();

  const int wv = tid >> 5;
  const int lane = tid & 31;
  const int gw = blockIdx.x * 8 + wv;
  const int nw = gridDim.x * 8;
  const float b2 = b2p[0];

  for (int tile = gw; tile < NTOT / 16; tile += nw) {
    for (int t = 0; t < 16; ++t) {
      int gid = tile * 16 + t;
      int s = gid / NNODE;
      int n = gid - s * NNODE;
      float mean, mn, mx, sd;
      pna_stats(m, s, n, lane, mean, mn, mx, sd);
      float amp = c_logd[n] * (1.0f / AVG_LOG);
      float att = AVG_LOG / c_logd[n];
      float st[4] = {mean, mn, mx, sd};
      float sc[3] = {1.0f, amp, att};
      float pd = 0.f;
      #pragma unroll
      for (int g = 0; g < 3; ++g)
        #pragma unroll
        for (int q = 0; q < 4; ++q)
          pd += st[q] * sc[g] * sW[g * 128 + q * 32 + lane];
      // butterfly sum across 32 lanes
      for (int off = 16; off > 0; off >>= 1) pd += __shfl_xor(pd, off, 32);
      if (lane == 0) {
        float p = 1.0f / (1.0f + expf(-(pd + b2)));
        atomicAdd(&pmacc[n], p);
      }
    }
  }
  __syncthreads();
  if (tid < NNODE) atomicAdd(&pmout[tid], pmacc[tid]);
}

// ---------------- final: out[i] = mean_b(p) @ W_fc[i] + b_fc[i] -------------
__global__ void k_final(const float* __restrict__ pm, const float* __restrict__ Wfc,
                        const float* __restrict__ bfc, float* __restrict__ out) {
  int i = threadIdx.x;
  if (i < NINT) {
    float acc = bfc[i];
    for (int n = 0; n < NNODE; ++n)
      acc += (pm[i * NNODE + n] * (1.0f / (float)BBATCH)) * Wfc[i * NNODE + n];
    out[i] = acc;
  }
}

extern "C" void kernel_launch(void* const* d_in, const int* in_sizes, int n_in,
                              void* d_out, int out_size, void* d_ws, size_t ws_size,
                              hipStream_t stream) {
  const float* x      = (const float*)d_in[0];
  const float* eattr  = (const float*)d_in[1];
  const float* Wemb   = (const float*)d_in[2];
  const float* bemb   = (const float*)d_in[3];
  const float* Wpre1  = (const float*)d_in[4];
  const float* bpre1  = (const float*)d_in[5];
  const float* Wpost1 = (const float*)d_in[6];
  const float* bpost1 = (const float*)d_in[7];
  const float* Wpre2  = (const float*)d_in[8];
  const float* bpre2  = (const float*)d_in[9];
  const float* Wpost2 = (const float*)d_in[10];
  const float* bpost2 = (const float*)d_in[11];
  const float* Wfc    = (const float*)d_in[12];
  const float* bfc    = (const float*)d_in[13];

  char* ws = (char*)d_ws;
  float*     pm = (float*)ws;                                        // 3*11 f32
  _Float16*  H  = (_Float16*)(ws + 256);                             // NTOT*32 f16
  _Float16*  h1 = (_Float16*)(ws + 256 + (size_t)NTOT * FDIM * 2);   // NTOT*32 f16
  float*     m  = (float*)(ws + 256 + (size_t)NTOT * FDIM * 4);      // ETOT*32 f32

  hipMemsetAsync(pm, 0, NINT * NNODE * sizeof(float), stream);
  k_encoder<<<(NTOT * FDIM + 255) / 256, 256, 0, stream>>>(x, Wemb, bemb, H);

  for (int i = 0; i < NINT; ++i) {
    k_edge_pre<<<256, 128, 0, stream>>>(H, eattr, Wpre1 + (size_t)i * KPRE * FDIM,
                                        bpre1 + i * FDIM, m);
    k_node_post1<<<352, 64, 0, stream>>>(m, Wpost1 + (size_t)i * KPOST * FDIM,
                                         bpost1 + i * FDIM, h1);
    k_edge_pre<<<256, 128, 0, stream>>>(h1, eattr, Wpre2 + (size_t)i * KPRE * FDIM,
                                        bpre2 + i * FDIM, m);
    k_node_post2<<<128, 256, 0, stream>>>(m, Wpost2 + (size_t)i * KPOST,
                                          bpost2 + i, pm + i * NNODE);
  }
  k_final<<<1, 64, 0, stream>>>(pm, Wfc, bfc, (float*)d_out);
}
